// HybridQLSTM_65481071410450
// MI455X (gfx1250) — compile-verified, matrix operand order checked by reading, not run on hardware
//
#include <hip/hip_runtime.h>
#include <stdint.h>

// ---------------------------------------------------------------------------
// LSTM (T=2048, B=64, D_IN=D_H=256) for MI455X / gfx1250.
// Phase 0: pack Wx/Wh (f32) -> bf16 WMMA B-fragment layout in ws.
// Phase 1: time-parallel  Xg = X @ Wx + b  (bf16 WMMA); x[t] staged into LDS
//          pre-converted into A-fragment-packed bf16; output stored in
//          D-fragment-packed bf16 so phase 2 can bulk-copy it.
// Phase 2: persistent single-workgroup scan: gates = Xg[t] + h @ Wh (WMMA);
//          Xg[t] tile (128KB) async-copied into LDS with
//          global_load_async_to_lds_b128, overlapped with the cell update.
// ws: [0,512K) Wx-pack | [512K,1M) Wh-pack | [1M,+256MB) Xg (all bf16)
// ---------------------------------------------------------------------------

typedef __attribute__((ext_vector_type(16))) __bf16       v16bf;
typedef __attribute__((ext_vector_type(8)))  float        v8f;
typedef __attribute__((ext_vector_type(4)))  unsigned int u32x4;
typedef __attribute__((ext_vector_type(4)))  float        f32x4;

#define T_STEPS 2048
#define BATCH   64
#define DIN     256
#define DH      256
#define NG      1024   // 4*DH
#define KT      8      // 256 / 32

// B fragment (32x16 KxN), ISA 7.12.2: lane holds N = lane&15, hi = lane>>4;
// element e -> K = (e&7) + (e&8)*2 + hi*8.
// A fragment (16x32 MxK): lane holds M = lane&15, same K(e) mapping.
// C/D (16x16): VGPR r -> M = r + 8*hi, N = lane&15.

__global__ __launch_bounds__(256) void pack_weights(
    const float* __restrict__ Wf, const float* __restrict__ Wi,
    const float* __restrict__ Wg, const float* __restrict__ Wo,
    __bf16* __restrict__ wxp, __bf16* __restrict__ whp) {
  int tid  = blockIdx.x * blockDim.x + threadIdx.x;
  int e    = tid & 15;
  int lane = (tid >> 4) & 31;
  int kt   = (tid >> 9) & 7;
  int nt   = (tid >> 12) & 63;
  int sel  = (tid >> 18) & 1;                          // 0 = Wx, 1 = Wh
  int hi   = lane >> 4;
  int k    = kt * 32 + (e & 7) + ((e & 8) << 1) + hi * 8;
  int n    = nt * 16 + (lane & 15);
  int gate = n >> 8;                                   // 0=f 1=i 2=g 3=o
  int j    = n & 255;
  const float* W = (gate == 0) ? Wf : (gate == 1) ? Wi : (gate == 2) ? Wg : Wo;
  float v = W[j * (DIN + DH) + k + (sel ? DIN : 0)];
  __bf16* dst = sel ? whp : wxp;
  dst[tid & 0x3FFFF] = (__bf16)v;
}

// ---------------------------------------------------------------------------
// Phase 1: one block per timestep t. 8 waves, wave owns 8 ntiles (2 at a time).
// ---------------------------------------------------------------------------
__global__ __launch_bounds__(256) void xgemm(
    const float* __restrict__ X, const __bf16* __restrict__ wxp,
    const float* __restrict__ bf_, const float* __restrict__ bi_,
    const float* __restrict__ bg_, const float* __restrict__ bo_,
    __bf16* __restrict__ Xg) {
  // A-fragment-packed x[t]: frag (kt*4+mt), 32 lanes x 16 bf16 = 32KB total.
  __shared__ __bf16 xaPack[32 * 32 * 16];
  const int t   = blockIdx.x;
  const int tid = threadIdx.x;

  // Stage + convert + swizzle: 2048 groups of 8 consecutive f32.
  {
    const float* src = X + (size_t)t * BATCH * DIN;
    for (int i = 0; i < 8; ++i) {
      int g  = i * 256 + tid;
      int m  = g >> 5;           // batch row 0..63
      int kg = g & 31;           // 8-wide k group
      const f32x4* ps = (const f32x4*)(src + m * DIN + kg * 8);
      f32x4 lo = ps[0], hv = ps[1];
      int kt_  = kg >> 2;
      int khi  = kg & 1;                 // hi half-group select
      int e0   = 8 * ((kg >> 1) & 1);    // element offset 0 or 8
      int mt   = m >> 4;
      int lane_d = (m & 15) + 16 * khi;
      union { __bf16 h[8]; u32x4 q; } u;
      for (int j = 0; j < 4; ++j) { u.h[j] = (__bf16)lo[j]; u.h[4 + j] = (__bf16)hv[j]; }
      *(u32x4*)&xaPack[((kt_ * 4 + mt) * 32 + lane_d) * 16 + e0] = u.q;
    }
  }
  __syncthreads();

  const int wave = tid >> 5;
  const int lane = tid & 31;
  const int col  = lane & 15;

  for (int p = 0; p < 4; ++p) {          // wave's 8 ntiles, 2 per pass
    const int ntb = wave * 8 + p * 2;
    v8f acc[2][4];
    for (int q = 0; q < 2; ++q) {
      const int n = (ntb + q) * 16 + col;
      const int gate = n >> 8, j = n & 255;
      const float bval =
          (gate == 0) ? bf_[j] : (gate == 1) ? bi_[j] : (gate == 2) ? bg_[j] : bo_[j];
      for (int mt = 0; mt < 4; ++mt)
        for (int r = 0; r < 8; ++r) acc[q][mt][r] = bval;
    }
    for (int kt = 0; kt < KT; ++kt) {
      v16bf b0 = *(const v16bf*)(wxp + (((size_t)(ntb + 0) * KT + kt) * 32 + lane) * 16);
      v16bf b1 = *(const v16bf*)(wxp + (((size_t)(ntb + 1) * KT + kt) * 32 + lane) * 16);
      for (int mt = 0; mt < 4; ++mt) {
        union { v16bf v; u32x4 q[2]; } af;
        const __bf16* pa = &xaPack[((kt * 4 + mt) * 32 + lane) * 16];
        af.q[0] = *(const u32x4*)(pa);
        af.q[1] = *(const u32x4*)(pa + 8);
        acc[0][mt] = __builtin_amdgcn_wmma_f32_16x16x32_bf16(
            false, af.v, false, b0, (short)0, acc[0][mt], false, false);
        acc[1][mt] = __builtin_amdgcn_wmma_f32_16x16x32_bf16(
            false, af.v, false, b1, (short)0, acc[1][mt], false, false);
      }
    }
    // Store D-fragment-packed bf16: one b128 per tile per lane.
    for (int q = 0; q < 2; ++q)
      for (int mt = 0; mt < 4; ++mt) {
        union { __bf16 h[8]; u32x4 u; } o;
        for (int r = 0; r < 8; ++r) o.h[r] = (__bf16)acc[q][mt][r];
        size_t idx = ((((size_t)t * 64 + (ntb + q)) * 4 + mt) * 32 + lane) * 8;
        *(u32x4*)(Xg + idx) = o.u;
      }
  }
}

// ---------------------------------------------------------------------------
// Phase 2: persistent single workgroup (512 threads = 16 waves).
// LDS: gatesPack 128KB | xgLds 128KB | hLds 32KB  = 288KB.
// ---------------------------------------------------------------------------
__global__ __launch_bounds__(512) void lstm_scan(
    const __bf16* Xg, const __bf16* whp,
    const float* __restrict__ h0, const float* __restrict__ c0,
    float* out) {
  extern __shared__ char smem[];
  __bf16* gatesPack = (__bf16*)smem;                    // 64*4*32*8 = 65536 elems
  __bf16* xgLds     = (__bf16*)(smem + 131072);         // 65536 elems
  __bf16* hLds      = (__bf16*)(smem + 262144);         // 64*256 row-major

  const int tid  = threadIdx.x;
  const int wave = tid >> 5;
  const int lane = tid & 31;
  const int hi   = lane >> 4;
  const int col  = lane & 15;
  const uint64_t xgGlobal = (uint64_t)(uintptr_t)Xg;
  const uint32_t xgLdsOff = (uint32_t)(uintptr_t)(void*)xgLds;

  float c[32];
  for (int i = 0; i < 32; ++i) {
    int idx = i * 512 + tid;
    c[i]      = c0[idx];
    hLds[idx] = (__bf16)h0[idx];
  }
  // Async-copy Xg[0] tile (128KB) into LDS.
  for (int i = 0; i < 16; ++i) {
    uint32_t ch = (uint32_t)(i * 512 + tid);
    uint64_t ga = xgGlobal + (uint64_t)ch * 16;
    uint32_t la = xgLdsOff + ch * 16;
    asm volatile("global_load_async_to_lds_b128 %0, %1, off" :: "v"(la), "v"(ga) : "memory");
  }
  asm volatile("s_wait_asynccnt 0x0" ::: "memory");
  __syncthreads();

  for (int t = 0; t < T_STEPS; ++t) {
    // ---- gates = Xg[t] + h @ Wh : wave owns ntiles wave*4..+3, 2 per pass
    for (int p = 0; p < 2; ++p) {
      const int ntb = wave * 4 + p * 2;
      v8f acc[2][4];
      for (int q = 0; q < 2; ++q)
        for (int mt = 0; mt < 4; ++mt) {
          union { __bf16 h[8]; u32x4 u; } xu;
          xu.u = *(const u32x4*)&xgLds[(((ntb + q) * 4 + mt) * 32 + lane) * 8];
          for (int r = 0; r < 8; ++r) acc[q][mt][r] = (float)xu.h[r];
        }
      for (int kt = 0; kt < KT; ++kt) {
        v16bf b0 = *(const v16bf*)(whp + (((size_t)(ntb + 0) * KT + kt) * 32 + lane) * 16);
        v16bf b1 = *(const v16bf*)(whp + (((size_t)(ntb + 1) * KT + kt) * 32 + lane) * 16);
        for (int mt = 0; mt < 4; ++mt) {
          union { v16bf v; u32x4 q[2]; } af;
          const __bf16* pa = &hLds[(mt * 16 + col) * DH + kt * 32 + hi * 8];
          af.q[0] = *(const u32x4*)(pa);        // K = base .. base+7
          af.q[1] = *(const u32x4*)(pa + 16);   // K = base+16 .. base+23
          acc[0][mt] = __builtin_amdgcn_wmma_f32_16x16x32_bf16(
              false, af.v, false, b0, (short)0, acc[0][mt], false, false);
          acc[1][mt] = __builtin_amdgcn_wmma_f32_16x16x32_bf16(
              false, af.v, false, b1, (short)0, acc[1][mt], false, false);
        }
      }
      for (int q = 0; q < 2; ++q)
        for (int mt = 0; mt < 4; ++mt) {
          union { __bf16 h[8]; u32x4 u; } o;
          for (int r = 0; r < 8; ++r) o.h[r] = (__bf16)acc[q][mt][r];
          *(u32x4*)&gatesPack[(((ntb + q) * 4 + mt) * 32 + lane) * 8] = o.u;
        }
    }
    __syncthreads();

    // ---- prefetch Xg[t+1] into xgLds; lands during the cell update
    if (t + 1 < T_STEPS) {
      uint64_t base = xgGlobal + (uint64_t)(t + 1) * 131072u;
      for (int i = 0; i < 16; ++i) {
        uint32_t ch = (uint32_t)(i * 512 + tid);
        uint64_t ga = base + (uint64_t)ch * 16;
        uint32_t la = xgLdsOff + ch * 16;
        asm volatile("global_load_async_to_lds_b128 %0, %1, off" :: "v"(la), "v"(ga) : "memory");
      }
    }

    // ---- cell update (f32); reads gatesPack in D-fragment layout
    for (int i = 0; i < 32; ++i) {
      int idx = i * 512 + tid;
      int b = idx >> 8, j = idx & 255;
      int mt = b >> 4, r = b & 7, hb = (b >> 3) & 1;
      // pre-activation at column n: offset ((n>>4)*4+mt)*32 + (n&15)+16*hb, elem r
      int jc = j & 15;
      float fpre = (float)gatesPack[((((0 * 256 + j) >> 4) * 4 + mt) * 32 + jc + 16 * hb) * 8 + r];
      float ipre = (float)gatesPack[((((256 + j) >> 4) * 4 + mt) * 32 + jc + 16 * hb) * 8 + r];
      float gpre = (float)gatesPack[((((512 + j) >> 4) * 4 + mt) * 32 + jc + 16 * hb) * 8 + r];
      float opre = (float)gatesPack[((((768 + j) >> 4) * 4 + mt) * 32 + jc + 16 * hb) * 8 + r];
      float fg = 1.f / (1.f + __expf(-fpre));
      float ig = 1.f / (1.f + __expf(-ipre));
      float gg = tanhf(gpre);
      float og = 1.f / (1.f + __expf(-opre));
      float cn = fg * c[i] + ig * gg;
      c[i] = cn;
      float hn = og * tanhf(cn);
      out[((size_t)t * BATCH + b) * DH + j] = hn;
      hLds[idx] = (__bf16)hn;
      if (t == T_STEPS - 1) {
        out[(size_t)T_STEPS * BATCH * DH + idx]              = hn;  // hx
        out[(size_t)T_STEPS * BATCH * DH + BATCH * DH + idx] = cn;  // cx
      }
    }
    asm volatile("s_wait_asynccnt 0x0" ::: "memory");
    __syncthreads();
  }
}

// ---------------------------------------------------------------------------
extern "C" void kernel_launch(void* const* d_in, const int* in_sizes, int n_in,
                              void* d_out, int out_size, void* d_ws, size_t ws_size,
                              hipStream_t stream) {
  const float* X   = (const float*)d_in[0];
  const float* h0  = (const float*)d_in[1];
  const float* c0  = (const float*)d_in[2];
  const float* Wf  = (const float*)d_in[3];
  const float* Wi  = (const float*)d_in[4];
  const float* Wg  = (const float*)d_in[5];
  const float* Wo  = (const float*)d_in[6];
  const float* bf_ = (const float*)d_in[7];
  const float* bi_ = (const float*)d_in[8];
  const float* bg_ = (const float*)d_in[9];
  const float* bo_ = (const float*)d_in[10];
  float* out = (float*)d_out;

  __bf16* wxp = (__bf16*)d_ws;          // 262144 elems (512 KB)
  __bf16* whp = wxp + 262144;           // 262144 elems (512 KB)
  __bf16* Xg  = whp + 262144;           // 2048*65536 elems (256 MB), frag-packed

  pack_weights<<<2048, 256, 0, stream>>>(Wf, Wi, Wg, Wo, wxp, whp);
  xgemm<<<T_STEPS, 256, 0, stream>>>(X, wxp, bf_, bi_, bg_, bo_, Xg);
  size_t smem = 131072 + 131072 + 32768;                // 288 KB
  lstm_scan<<<1, 512, smem, stream>>>(Xg, whp, h0, c0, out);
}
